// LlamaAttention_22728966930907
// MI455X (gfx1250) — matrix operand order repaired and checked by reading, not compiled, not run on hardware
//
#include <hip/hip_runtime.h>
#include <hip/hip_bf16.h>

// ---------------------------------------------------------------------------
// Problem constants (from reference)
// ---------------------------------------------------------------------------
constexpr int Bc    = 2;
constexpr int Sc    = 2048;
constexpr int HIDc  = 2048;
constexpr int Hc    = 16;
constexpr int Dc    = 128;
constexpr int QSIZE = Hc * Dc;        // 2048
constexpr int COREc = 3 * QSIZE;      // 6144
constexpr int ALLc  = COREc + 2 * Hc; // 6176
constexpr float SCALEc = 0.08838834764831845f; // 128^-0.5

// ---------------------------------------------------------------------------
// Vector types
// ---------------------------------------------------------------------------
typedef __attribute__((ext_vector_type(16))) __bf16 v16bf;
typedef __attribute__((ext_vector_type(8)))  __bf16 v8bf;
typedef __attribute__((ext_vector_type(8)))  float  v8f;
typedef __attribute__((ext_vector_type(4)))  float  f32x4;
typedef __attribute__((ext_vector_type(4)))  unsigned int u32x4;
typedef __attribute__((ext_vector_type(8)))  int    i32x8;
typedef __attribute__((ext_vector_type(4)))  int    i32x4;

static __device__ inline v16bf load16_bf16(const float* p) {
  f32x4 x0 = *(const f32x4*)(p);
  f32x4 x1 = *(const f32x4*)(p + 4);
  f32x4 x2 = *(const f32x4*)(p + 8);
  f32x4 x3 = *(const f32x4*)(p + 12);
  v16bf r;
#pragma unroll
  for (int i = 0; i < 4; ++i) {
    r[i]      = (__bf16)x0[i];
    r[4 + i]  = (__bf16)x1[i];
    r[8 + i]  = (__bf16)x2[i];
    r[12 + i] = (__bf16)x3[i];
  }
  return r;
}

// A-fragment for 16-bit WMMA: two contiguous 8-float segments -> elts 0..7, 8..15
static __device__ inline v16bf load8x2_bf16(const float* p0, const float* p1) {
  f32x4 x0 = *(const f32x4*)(p0);
  f32x4 x1 = *(const f32x4*)(p0 + 4);
  f32x4 x2 = *(const f32x4*)(p1);
  f32x4 x3 = *(const f32x4*)(p1 + 4);
  v16bf r;
#pragma unroll
  for (int i = 0; i < 4; ++i) {
    r[i]      = (__bf16)x0[i];
    r[4 + i]  = (__bf16)x1[i];
    r[8 + i]  = (__bf16)x2[i];
    r[12 + i] = (__bf16)x3[i];
  }
  return r;
}

static __device__ inline float softplus_f(float x) {
  return fmaxf(x, 0.0f) + log1pf(__expf(-fabsf(x)));
}

// ---------------------------------------------------------------------------
// Tensor Data Mover: 2D tile load (f32 elements) into LDS.
// D# built per CDNA5 ISA 8.3/8.4: group0 {count=1, lds_addr, global_addr,
// type=2}; group1 {data_size=4B, tensor_dim0/1 (OOB -> zero fill),
// tile_dim0/1, tensor_dim0_stride}. Groups 2/3/4 unused (2D tensor).
// This toolchain exposes the 6-arg clang-23 builtin form.
// ---------------------------------------------------------------------------
static __device__ inline void tdm_load_2d_f32(
    unsigned lds_addr, const float* gptr,
    unsigned tensor_d0, unsigned tensor_d1,   // full tensor extents (elements)
    unsigned tile_d0,   unsigned tile_d1,     // tile extents (elements)
    unsigned stride_d0)                       // row stride (elements)
{
  unsigned long long ga = (unsigned long long)(uintptr_t)gptr;
  u32x4 g0;
  g0[0] = 1u;                                            // count=1, user D#
  g0[1] = lds_addr;                                      // lds_addr (bytes)
  g0[2] = (unsigned)(ga & 0xFFFFFFFFu);                  // global_addr[31:0]
  g0[3] = (unsigned)((ga >> 32) & 0x01FFFFFFu) | (2u << 30);  // [56:32] | type=2
  i32x8 g1;
  g1[0] = (int)(2u << 16);                               // data_size=4B, wg_mask=0
  g1[1] = (int)((tensor_d0 & 0xFFFFu) << 16);            // tensor_dim0[15:0]
  g1[2] = (int)(((tensor_d0 >> 16) & 0xFFFFu) |          // tensor_dim0[31:16]
                ((tensor_d1 & 0xFFFFu) << 16));          // tensor_dim1[15:0]
  g1[3] = (int)(((tensor_d1 >> 16) & 0xFFFFu) |          // tensor_dim1[31:16]
                ((tile_d0 & 0xFFFFu) << 16));            // tile_dim0
  g1[4] = (int)(tile_d1 & 0xFFFFu);                      // tile_dim1 (tile_dim2=0)
  g1[5] = (int)stride_d0;                                // dim0_stride[31:0]
  g1[6] = 0;                                             // dim0_stride[47:32]=0
  g1[7] = 0;
  i32x4 g2 = {0, 0, 0, 0};
  i32x4 g3 = {0, 0, 0, 0};
  i32x8 g4 = {0, 0, 0, 0, 0, 0, 0, 0};
  __builtin_amdgcn_tensor_load_to_lds(g0, g1, g2, g3, g4, 0);
}

static __device__ inline unsigned lds_byte_off(const void* p) {
  return (unsigned)(uintptr_t)p;   // generic LDS addr: low 32 bits = LDS offset
}

// ---------------------------------------------------------------------------
// GEMM: C[M,N] = A[M,K] * W[N,K]^T   (f32 memory, bf16 WMMA math)
// Block = 8 waves -> 64(M) x 128(N) macro-tile; wave tile = 32x32 (2x2 WMMA).
// K-loop in chunks of 32, double-buffered LDS fed by the Tensor Data Mover
// (wave 0 issues tensor_load_to_lds; TENSORcnt + workgroup barrier sync).
// WMMA lane layouts (wave32, ISA 7.12.2):
//   A 16x32 : lane L row = L%16; elts 0..7 -> K=(L<16?0:8)+i, 8..15 -> +16
//   B 32x16 : lane L col = L%16; elts 0..7 -> K=(L<16?0:8)+i, 8..15 -> +16
//   C 16x16 : lane L col = L%16; acc[p] -> row M = p + 8*(L/16)
// ---------------------------------------------------------------------------
__global__ __launch_bounds__(256) void gemm_bt_tdm(
    const float* __restrict__ A,  // [M,K]
    const float* __restrict__ W,  // [N,K]
    float* __restrict__ C,        // [M,N]
    int M, int N, int K) {
  __shared__ __attribute__((aligned(16))) float lA[2][64][32];
  __shared__ __attribute__((aligned(16))) float lW[2][128][32];

  const int lane = threadIdx.x & 31;
  const int widx = threadIdx.x >> 5;
  const int Nb   = (N + 127) >> 7;
  const int bM   = blockIdx.x / Nb;
  const int bN   = blockIdx.x - bM * Nb;
  const int row0 = bM * 64;
  const int col0 = bN * 128;
  const int wm   = widx >> 2;      // 0..1
  const int wn   = widx & 3;       // 0..3
  const int r    = lane & 15;
  const int half = lane >> 4;

  const int wrow  = wm * 32;       // row base inside lA
  const int wcol  = wn * 32;       // row base inside lW (C column)
  const int cn0   = col0 + wcol;   // global column base of this wave

  v8f acc[2][2];
#pragma unroll
  for (int i = 0; i < 2; ++i)
#pragma unroll
    for (int j = 0; j < 2; ++j)
      acc[i][j] = (v8f){0.f, 0.f, 0.f, 0.f, 0.f, 0.f, 0.f, 0.f};

  const int niter = K >> 5;
  if (widx == 0) {  // prologue: DMA chunk 0 into buffer 0
    tdm_load_2d_f32(lds_byte_off(&lA[0][0][0]), A + (size_t)row0 * K,
                    (unsigned)K, (unsigned)(M - row0), 32, 64, (unsigned)K);
    tdm_load_2d_f32(lds_byte_off(&lW[0][0][0]), W + (size_t)col0 * K,
                    (unsigned)K, (unsigned)(N - col0), 32, 128, (unsigned)K);
  }

  for (int it = 0; it < niter; ++it) {
    const int cur = it & 1;
    if (widx == 0) {
      if (it + 1 < niter) {   // DMA next chunk into the other buffer
        const int k1 = (it + 1) << 5;
        tdm_load_2d_f32(lds_byte_off(&lA[cur ^ 1][0][0]),
                        A + (size_t)row0 * K + k1,
                        (unsigned)(K - k1), (unsigned)(M - row0), 32, 64,
                        (unsigned)K);
        tdm_load_2d_f32(lds_byte_off(&lW[cur ^ 1][0][0]),
                        W + (size_t)col0 * K + k1,
                        (unsigned)(K - k1), (unsigned)(N - col0), 32, 128,
                        (unsigned)K);
        __builtin_amdgcn_s_wait_tensorcnt(2);  // chunk `it` complete
      } else {
        __builtin_amdgcn_s_wait_tensorcnt(0);
      }
    }
    __syncthreads();  // LDS chunk `it` visible to all 8 waves

    v16bf a0 = load8x2_bf16(&lA[cur][wrow + r][half * 8],
                            &lA[cur][wrow + r][half * 8 + 16]);
    v16bf a1 = load8x2_bf16(&lA[cur][wrow + 16 + r][half * 8],
                            &lA[cur][wrow + 16 + r][half * 8 + 16]);
    v16bf b0 = load8x2_bf16(&lW[cur][wcol + r][half * 8],
                            &lW[cur][wcol + r][half * 8 + 16]);
    v16bf b1 = load8x2_bf16(&lW[cur][wcol + 16 + r][half * 8],
                            &lW[cur][wcol + 16 + r][half * 8 + 16]);

    acc[0][0] = __builtin_amdgcn_wmma_f32_16x16x32_bf16(
        false, a0, false, b0, (short)0, acc[0][0], false, false);
    acc[0][1] = __builtin_amdgcn_wmma_f32_16x16x32_bf16(
        false, a0, false, b1, (short)0, acc[0][1], false, false);
    acc[1][0] = __builtin_amdgcn_wmma_f32_16x16x32_bf16(
        false, a1, false, b0, (short)0, acc[1][0], false, false);
    acc[1][1] = __builtin_amdgcn_wmma_f32_16x16x32_bf16(
        false, a1, false, b1, (short)0, acc[1][1], false, false);

    __syncthreads();  // done reading before buffer is overwritten
  }

  // store 2x2 tiles; guard the (partial) last N-block columns
#pragma unroll
  for (int i = 0; i < 2; ++i) {
#pragma unroll
    for (int j = 0; j < 2; ++j) {
      const int col = cn0 + j * 16 + r;
      if (col < N) {
        float* Crow =
            C + (size_t)(row0 + wrow + i * 16 + 8 * half) * N + col;
#pragma unroll
        for (int p = 0; p < 8; ++p) Crow[(size_t)p * N] = acc[i][j][p];
      }
    }
  }
}

// ---------------------------------------------------------------------------
// Per (b,h) sequential scan: cumA = cumsum(-softplus), dt = softplus.
// ---------------------------------------------------------------------------
__global__ void scan_cumA_dt(const float* __restrict__ qkv,
                             float* __restrict__ cumA,
                             float* __restrict__ dtb) {
  const int t = blockIdx.x * blockDim.x + threadIdx.x;
  if (t >= Bc * Hc) return;
  const int b = t / Hc, h = t - b * Hc;
  const float* base = qkv + (size_t)b * Sc * ALLc;
  float* ca = cumA + ((size_t)b * Hc + h) * Sc;
  float* dv = dtb  + ((size_t)b * Hc + h) * Sc;
  float run = 0.f;
  for (int s = 0; s < Sc; ++s) {
    const float* row = base + (size_t)s * ALLc;
    run += -softplus_f(row[COREc + h]);
    ca[s] = run;
    dv[s] = softplus_f(row[COREc + Hc + h]);
  }
}

// ---------------------------------------------------------------------------
// 2-tap causal depthwise conv over sequence.
// ---------------------------------------------------------------------------
__global__ void conv2tap(const float* __restrict__ qkv,
                         const float* __restrict__ conv_w,  // [CORE,2]
                         const float* __restrict__ conv_b,  // [CORE]
                         float* __restrict__ core) {
  const size_t total = (size_t)Bc * Sc * COREc;
  size_t i = (size_t)blockIdx.x * blockDim.x + threadIdx.x;
  if (i >= total) return;
  const int c = (int)(i % COREc);
  const size_t bs = i / COREc;
  const int s = (int)(bs % Sc);
  const float cur  = qkv[bs * ALLc + c];
  const float prev = (s == 0) ? 0.f : qkv[(bs - 1) * ALLc + c];
  core[i] = conv_w[2 * c] * prev + conv_w[2 * c + 1] * cur + conv_b[c];
}

// ---------------------------------------------------------------------------
// Decayed squared-score attention (flash style).
// One wave per (b, h, 16-row query block); key loop in 32-key steps.
// LDS re-layouts the 16x32 attention tile from C-layout to A-layout.
// ---------------------------------------------------------------------------
__global__ __launch_bounds__(256) void attn_decay(
    const float* __restrict__ core,   // [B,S,CORE]
    const float* __restrict__ cumA,   // [B,H,S]
    const float* __restrict__ dtb,    // [B,H,S]
    float* __restrict__ outp) {       // [B,S,QSIZE]
  __shared__ __attribute__((aligned(16))) __bf16 atile[8][16][32];

  const int lane = threadIdx.x & 31;
  const int widx = threadIdx.x >> 5;
  const int QB   = Sc / 16;
  const int wv   = blockIdx.x * 8 + widx;
  if (wv >= Bc * Hc * QB) return;
  const int qb = wv % QB;
  const int h  = (wv / QB) % Hc;
  const int b  = wv / (QB * Hc);
  const int n    = lane & 15;
  const int half = lane >> 4;
  const int qbase = qb * 16;

  const float* cumAbh = cumA + ((size_t)b * Hc + h) * Sc;
  const float* dtbh   = dtb  + ((size_t)b * Hc + h) * Sc;
  const float* qrow   = core + ((size_t)b * Sc + qbase + n) * COREc + h * Dc;
  const float* kbase  = core + (size_t)b * Sc * COREc + QSIZE     + h * Dc;
  const float* vbase  = core + (size_t)b * Sc * COREc + 2 * QSIZE + h * Dc;

  v16bf qf[4];
#pragma unroll
  for (int c = 0; c < 4; ++c)
    qf[c] = load8x2_bf16(qrow + c * 32 + half * 8,
                         qrow + c * 32 + half * 8 + 16);

  float cqa[8];
#pragma unroll
  for (int e = 0; e < 8; ++e) cqa[e] = cumAbh[qbase + e + 8 * half];

  v8f num[8];
#pragma unroll
  for (int t = 0; t < 8; ++t) num[t] = (v8f){0.f,0.f,0.f,0.f,0.f,0.f,0.f,0.f};
  float den[8] = {0.f,0.f,0.f,0.f,0.f,0.f,0.f,0.f};

  const int nb32 = qb / 2 + 1;
  for (int jb = 0; jb < nb32; ++jb) {
    const int j0 = jb * 32;
    v8f s0 = (v8f){0.f,0.f,0.f,0.f,0.f,0.f,0.f,0.f};
    v8f s1 = (v8f){0.f,0.f,0.f,0.f,0.f,0.f,0.f,0.f};
#pragma unroll
    for (int c = 0; c < 4; ++c) {
      const float* kr0 = kbase + (size_t)(j0 + n) * COREc + c * 32 + half * 8;
      const float* kr1 = kbase + (size_t)(j0 + 16 + n) * COREc + c * 32 + half * 8;
      v16bf k0 = load8x2_bf16(kr0, kr0 + 16);
      s0 = __builtin_amdgcn_wmma_f32_16x16x32_bf16(
          false, qf[c], false, k0, (short)0, s0, false, false);
      v16bf k1 = load8x2_bf16(kr1, kr1 + 16);
      s1 = __builtin_amdgcn_wmma_f32_16x16x32_bf16(
          false, qf[c], false, k1, (short)0, s1, false, false);
    }
    const float cka0 = cumAbh[j0 + n];
    const float cka1 = cumAbh[j0 + 16 + n];
#pragma unroll
    for (int e = 0; e < 8; ++e) {
      const int m  = e + 8 * half;
      const int qi = qbase + m;
      float sc0 = s0[e] * SCALEc;
      float a0  = (j0 + n <= qi) ? sc0 * sc0 * __expf(cqa[e] - cka0) : 0.f;
      float sc1 = s1[e] * SCALEc;
      float a1  = (j0 + 16 + n <= qi) ? sc1 * sc1 * __expf(cqa[e] - cka1) : 0.f;
      den[e] += a0 + a1;
      atile[widx][m][n]      = (__bf16)a0;
      atile[widx][m][16 + n] = (__bf16)a1;
    }
    __asm__ volatile("" ::: "memory");  // keep DS store->load ordering

    v8bf p0 = *(const v8bf*)&atile[widx][n][half * 8];
    v8bf p1 = *(const v8bf*)&atile[widx][n][half * 8 + 16];
    v16bf af;
#pragma unroll
    for (int i = 0; i < 8; ++i) { af[i] = p0[i]; af[8 + i] = p1[i]; }

    float dts[16];
#pragma unroll
    for (int e = 0; e < 16; ++e) dts[e] = dtbh[j0 + ((e < 8) ? (half * 8 + e) : (16 + half * 8 + e - 8))];
#pragma unroll
    for (int t = 0; t < 8; ++t) {
      v16bf vf;
#pragma unroll
      for (int e = 0; e < 16; ++e) {
        float vv = vbase[(size_t)(j0 + ((e < 8) ? (half * 8 + e) : (16 + half * 8 + e - 8))) * COREc + t * 16 + n];
        vf[e] = (__bf16)(vv * dts[e]);
      }
      num[t] = __builtin_amdgcn_wmma_f32_16x16x32_bf16(
          false, af, false, vf, (short)0, num[t], false, false);
    }
  }

#pragma unroll
  for (int e = 0; e < 8; ++e) {
#pragma unroll
    for (int off = 8; off >= 1; off >>= 1)
      den[e] += __shfl_xor(den[e], off, 16);
    den[e] = 1.0f / den[e];
  }

#pragma unroll
  for (int t = 0; t < 8; ++t) {
#pragma unroll
    for (int e = 0; e < 8; ++e) {
      const int m = e + 8 * half;
      outp[((size_t)b * Sc + qbase + m) * QSIZE + h * Dc + t * 16 + n] =
          num[t][e] * den[e];
    }
  }
}

// ---------------------------------------------------------------------------
// Launch
// ---------------------------------------------------------------------------
extern "C" void kernel_launch(void* const* d_in, const int* in_sizes, int n_in,
                              void* d_out, int out_size, void* d_ws, size_t ws_size,
                              hipStream_t stream) {
  const float* hs     = (const float*)d_in[0];  // [B,S,HID]
  const float* w_qkv  = (const float*)d_in[1];  // [ALL,HID]
  const float* conv_w = (const float*)d_in[2];  // [CORE,2]
  const float* conv_b = (const float*)d_in[3];  // [CORE]
  const float* w_o    = (const float*)d_in[4];  // [HID,QSIZE]
  float* out = (float*)d_out;                   // [B,S,HID]

  float* qkv     = (float*)d_ws;                            // B*S*ALL
  float* corebuf = qkv     + (size_t)Bc * Sc * ALLc;        // B*S*CORE
  float* cumA    = corebuf + (size_t)Bc * Sc * COREc;       // B*H*S
  float* dtb     = cumA    + (size_t)Bc * Hc * Sc;          // B*H*S
  float* attnout = dtb     + (size_t)Bc * Hc * Sc;          // B*S*QSIZE

  // 1) qkv = hs @ w_qkv^T   (M=B*S, N=ALL, K=HID)
  {
    const int M = Bc * Sc, N = ALLc, K = HIDc;
    const int blocks = (M / 64) * ((N + 127) / 128);
    gemm_bt_tdm<<<blocks, 256, 0, stream>>>(hs, w_qkv, qkv, M, N, K);
  }
  // 2) cumA / dt scans
  scan_cumA_dt<<<1, 32, 0, stream>>>(qkv, cumA, dtb);
  // 3) causal 2-tap conv
  {
    const size_t total = (size_t)Bc * Sc * COREc;
    conv2tap<<<(unsigned)((total + 255) / 256), 256, 0, stream>>>(
        qkv, conv_w, conv_b, corebuf);
  }
  // 4) attention
  {
    const int waves = Bc * Hc * (Sc / 16);
    attn_decay<<<(waves + 7) / 8, 256, 0, stream>>>(corebuf, cumA, dtb, attnout);
  }
  // 5) out = attnout @ w_o^T  (M=B*S, N=HID, K=QSIZE)
  {
    const int M = Bc * Sc, N = HIDc, K = QSIZE;
    const int blocks = (M / 64) * ((N + 127) / 128);
    gemm_bt_tdm<<<blocks, 256, 0, stream>>>(attnout, w_o, out, M, N, K);
  }
}
